// ModularAttention_31602369364147
// MI455X (gfx1250) — matrix operand order repaired and checked by reading, not compiled, not run on hardware
//
#include <hip/hip_runtime.h>

#define SEQ 8192
#define DM  512
#define NH  8
#define DKH 64
#define KLD 72   // padded LDS leading dim (halves): 144B rows -> conflict-free b128 reads

typedef _Float16 v8h  __attribute__((ext_vector_type(8)));
typedef _Float16 v16h __attribute__((ext_vector_type(16)));
typedef float    v8f  __attribute__((ext_vector_type(8)));
typedef int      v4i  __attribute__((ext_vector_type(4)));

static __device__ __forceinline__ v16h cat8(v8h lo, v8h hi) {
  return __builtin_shufflevector(lo, hi, 0,1,2,3,4,5,6,7,8,9,10,11,12,13,14,15);
}

// A-operand fragment: 16x32 f16 tile from row-major memory (leading dim ld).
// Lane L<16: row L, K halves {0..7,16..23}; lane L>=16: row L-16, K {8..15,24..31}.
static __device__ __forceinline__ v16h load_fragA(const _Float16* __restrict__ base,
                                                  int ld, int lane) {
  const int r  = lane & 15;
  const int kh = (lane >> 4) << 3;
  const _Float16* p = base + (size_t)r * ld + kh;
  return cat8(*reinterpret_cast<const v8h*>(p), *reinterpret_cast<const v8h*>(p + 16));
}

// B-operand fragment: 32x16 f16 tile; memory row c (contiguous in K) supplies
// B column c. Lane L<16: col L, K {0..7,16..23}; lane L>=16: col L-16, K {8..15,24..31}.
static __device__ __forceinline__ v16h load_fragB(const _Float16* __restrict__ base,
                                                  int ld, int lane) {
  const int c  = lane & 15;
  const int kh = (lane >> 4) << 3;
  const _Float16* p = base + (size_t)c * ld + kh;
  return cat8(*reinterpret_cast<const v8h*>(p), *reinterpret_cast<const v8h*>(p + 16));
}

#define WMMA(a, b, c) \
  __builtin_amdgcn_wmma_f32_16x16x32_f16(false, (a), false, (b), (short)0, (c), false, false)

static __device__ __forceinline__ float hmax16(float v) {
  v = fmaxf(v, __shfl_xor(v, 1, 32));
  v = fmaxf(v, __shfl_xor(v, 2, 32));
  v = fmaxf(v, __shfl_xor(v, 4, 32));
  v = fmaxf(v, __shfl_xor(v, 8, 32));
  return v;
}
static __device__ __forceinline__ float hsum16(float v) {
  v += __shfl_xor(v, 1, 32);
  v += __shfl_xor(v, 2, 32);
  v += __shfl_xor(v, 4, 32);
  v += __shfl_xor(v, 8, 32);
  return v;
}

// ---- async global->LDS staging (CDNA5 ASYNCcnt path), guarded fallback ----
#if __has_builtin(__builtin_amdgcn_global_load_async_to_lds_b128)
#define ASYNC_CP 1
#else
#define ASYNC_CP 0
#endif

static __device__ __forceinline__ void cp16(_Float16* l, const _Float16* g) {
#if ASYNC_CP
  __attribute__((address_space(1))) v4i* gp =
      (__attribute__((address_space(1))) v4i*)(const __attribute__((address_space(1))) void*)g;
  __attribute__((address_space(3))) v4i* lp =
      (__attribute__((address_space(3))) v4i*)(__attribute__((address_space(3))) void*)l;
  __builtin_amdgcn_global_load_async_to_lds_b128(gp, lp, 0, 0);
#else
  *reinterpret_cast<v8h*>(l) = *reinterpret_cast<const v8h*>(g);
#endif
}

static __device__ __forceinline__ void async_wait_all() {
#if ASYNC_CP
#if __has_builtin(__builtin_amdgcn_s_wait_asynccnt)
  __builtin_amdgcn_s_wait_asynccnt(0);
#else
  asm volatile("s_wait_asynccnt 0" ::: "memory");
#endif
#endif
}

// ---------------------------------------------------------------- convert
__global__ __launch_bounds__(256) void cvt_f32_f16_kernel(
    const float* __restrict__ src, _Float16* __restrict__ dst, int n) {
  int i = blockIdx.x * 256 + threadIdx.x;
  if (i < n) dst[i] = (_Float16)src[i];
}

// ------------------------------------------------- QKV projection (WMMA GEMM)
// vt_mode==0: out[h][s][f] (Q/K, head-major rows); vt_mode==1: out[h][f][s] (V^T)
__global__ __launch_bounds__(256) void proj_qkv_kernel(
    const _Float16* __restrict__ xh, const _Float16* __restrict__ wh,
    const float* __restrict__ bias, _Float16* __restrict__ out, int vt_mode) {
  const int lane = threadIdx.x & 31;
  const int wave = threadIdx.x >> 5;
  const int m0 = blockIdx.x * 64  + (wave >> 2) * 32;
  const int n0 = blockIdx.y * 128 + (wave & 3) * 32;

  v8f acc[2][2] = {};
  for (int k0 = 0; k0 < DM; k0 += 32) {
    v16h a0 = load_fragA(xh + (size_t)(m0     ) * DM + k0, DM, lane);
    v16h a1 = load_fragA(xh + (size_t)(m0 + 16) * DM + k0, DM, lane);
    v16h b0 = load_fragB(wh + (size_t)(n0     ) * DM + k0, DM, lane);
    v16h b1 = load_fragB(wh + (size_t)(n0 + 16) * DM + k0, DM, lane);
    acc[0][0] = WMMA(a0, b0, acc[0][0]);
    acc[0][1] = WMMA(a0, b1, acc[0][1]);
    acc[1][0] = WMMA(a1, b0, acc[1][0]);
    acc[1][1] = WMMA(a1, b1, acc[1][1]);
  }

  const int col = lane & 15;
  const int rb  = (lane >> 4) << 3;
#pragma unroll
  for (int mt = 0; mt < 2; ++mt)
#pragma unroll
    for (int nt = 0; nt < 2; ++nt) {
      const int gc = n0 + nt * 16 + col;
      const float bb = bias[gc];
      const int h = gc >> 6;
      const int f = gc & 63;
      if (vt_mode) {
        v8h vrow;
#pragma unroll
        for (int i = 0; i < 8; ++i) vrow[i] = (_Float16)(acc[mt][nt][i] + bb);
        *reinterpret_cast<v8h*>(out + ((size_t)h * DKH + f) * SEQ + (m0 + mt * 16 + rb)) = vrow;
      } else {
#pragma unroll
        for (int i = 0; i < 8; ++i) {
          const int s = m0 + mt * 16 + rb + i;
          out[((size_t)h * SEQ + s) * DKH + f] = (_Float16)(acc[mt][nt][i] + bb);
        }
      }
    }
}

// ------------------------------------------------------- flash attention core
// Block = 8 waves x 16 query rows (128 rows), one head. Keys in blocks of 64,
// staged once per block into double-buffered LDS via async copies.
__global__ __launch_bounds__(256) void attn_kernel(
    const _Float16* __restrict__ qg, const _Float16* __restrict__ kg,
    const _Float16* __restrict__ vtg, _Float16* __restrict__ oc) {
  __shared__ _Float16 kbuf[2][64 * KLD];   // 64 keys x 64 feats (padded)
  __shared__ _Float16 vbuf[2][64 * KLD];   // 64 feats x 64 keys (padded)
  __shared__ _Float16 pbuf[8][16 * KLD];   // per-wave 16x64 P tile (padded)

  const int tid  = threadIdx.x;
  const int lane = tid & 31;
  const int wave = tid >> 5;
  const int h  = blockIdx.y;
  const int q0 = blockIdx.x * 128 + wave * 16;

  const _Float16* __restrict__ Q  = qg  + (size_t)h * SEQ * DKH;
  const _Float16* __restrict__ K  = kg  + (size_t)h * SEQ * DKH;
  const _Float16* __restrict__ VT = vtg + (size_t)h * DKH * SEQ;
  _Float16* pb = &pbuf[wave][0];

  // Q fragments, pre-scaled by 1/sqrt(dk) * log2(e) (exp2-domain softmax)
  v16h qa0 = load_fragA(Q + (size_t)q0 * DKH,      DKH, lane);
  v16h qa1 = load_fragA(Q + (size_t)q0 * DKH + 32, DKH, lane);
  {
    const _Float16 sch = (_Float16)(0.125f * 1.44269504089f);
#pragma unroll
    for (int i = 0; i < 16; ++i) { qa0[i] *= sch; qa1[i] *= sch; }
  }

  v8f o0 = {}, o1 = {}, o2 = {}, o3 = {};
  float mrow[8], lrow[8];
#pragma unroll
  for (int i = 0; i < 8; ++i) { mrow[i] = -3.0e38f; lrow[i] = 0.0f; }

  const int r  = lane & 15;
  const int rb = (lane >> 4) << 3;

  // cooperative staging: 256 threads x (2+2) x 16B = full 64x64 K and V tiles
  const int srow = tid >> 3;           // 0..31
  const int schk = (tid & 7) << 3;     // half-offset 0,8,...,56
  auto stage = [&](int buf, int j0) {
    cp16(&kbuf[buf][(size_t)(srow     ) * KLD + schk], K  + (size_t)(j0 + srow     ) * DKH + schk);
    cp16(&kbuf[buf][(size_t)(srow + 32) * KLD + schk], K  + (size_t)(j0 + srow + 32) * DKH + schk);
    cp16(&vbuf[buf][(size_t)(srow     ) * KLD + schk], VT + (size_t)(srow     ) * SEQ + j0 + schk);
    cp16(&vbuf[buf][(size_t)(srow + 32) * KLD + schk], VT + (size_t)(srow + 32) * SEQ + j0 + schk);
  };

  stage(0, 0);
  for (int j0 = 0, b = 0; j0 < SEQ; j0 += 64, ++b) {
    const int cur = b & 1;
    async_wait_all();
    __syncthreads();                       // tiles for this block visible to all
    if (j0 + 64 < SEQ) stage(cur ^ 1, j0 + 64);
    if (wave == 0 && j0 + 128 < SEQ) {
      __builtin_prefetch(K + (size_t)(j0 + 128) * DKH, 0, 1);
      __builtin_prefetch(VT + (j0 + 128), 0, 1);
    }

    const _Float16* kt = &kbuf[cur][0];
    const _Float16* vt = &vbuf[cur][0];

    // scores: 4 x (16x16) tiles over 64 keys, K-dim 64 (2 wmma each)
    v8f s[4] = {};
#pragma unroll
    for (int t = 0; t < 4; ++t) {
      v16h bk0 = load_fragB(kt + (size_t)(t * 16) * KLD,      KLD, lane);
      v16h bk1 = load_fragB(kt + (size_t)(t * 16) * KLD + 32, KLD, lane);
      s[t] = WMMA(qa0, bk0, s[t]);
      s[t] = WMMA(qa1, bk1, s[t]);
    }

    // online softmax over 64 keys at once
    float p[4][8];
#pragma unroll
    for (int i = 0; i < 8; ++i) {
      float mx = fmaxf(fmaxf(s[0][i], s[1][i]), fmaxf(s[2][i], s[3][i]));
      mx = hmax16(mx);
      const float mn = fmaxf(mrow[i], mx);
      const float al = exp2f(mrow[i] - mn);
      float rs = 0.0f;
#pragma unroll
      for (int t = 0; t < 4; ++t) { p[t][i] = exp2f(s[t][i] - mn); rs += p[t][i]; }
      rs = hsum16(rs);
      lrow[i] = lrow[i] * al + rs;
      mrow[i] = mn;
      o0[i] *= al; o1[i] *= al; o2[i] *= al; o3[i] *= al;
    }

    // P: C-layout -> per-wave LDS tile -> two A-fragments (in-order DS)
#pragma unroll
    for (int i = 0; i < 8; ++i)
#pragma unroll
      for (int t = 0; t < 4; ++t)
        pb[(size_t)(rb + i) * KLD + t * 16 + r] = (_Float16)p[t][i];
    __builtin_amdgcn_wave_barrier();
    v16h pa0, pa1;
    {
      const _Float16* pp = pb + (size_t)r * KLD + ((lane >> 4) << 3);
      pa0 = cat8(*(const v8h*)(pp),      *(const v8h*)(pp + 16));
      pa1 = cat8(*(const v8h*)(pp + 32), *(const v8h*)(pp + 48));
    }
    __builtin_amdgcn_wave_barrier();

    // O += P @ V : 4 feature tiles x 2 key-halves
    o0 = WMMA(pa0, load_fragB(vt + (size_t) 0 * KLD,      KLD, lane), o0);
    o0 = WMMA(pa1, load_fragB(vt + (size_t) 0 * KLD + 32, KLD, lane), o0);
    o1 = WMMA(pa0, load_fragB(vt + (size_t)16 * KLD,      KLD, lane), o1);
    o1 = WMMA(pa1, load_fragB(vt + (size_t)16 * KLD + 32, KLD, lane), o1);
    o2 = WMMA(pa0, load_fragB(vt + (size_t)32 * KLD,      KLD, lane), o2);
    o2 = WMMA(pa1, load_fragB(vt + (size_t)32 * KLD + 32, KLD, lane), o2);
    o3 = WMMA(pa0, load_fragB(vt + (size_t)48 * KLD,      KLD, lane), o3);
    o3 = WMMA(pa1, load_fragB(vt + (size_t)48 * KLD + 32, KLD, lane), o3);
  }

  // epilogue: normalize and store concatenated heads [S][DM] in f16
#pragma unroll
  for (int i = 0; i < 8; ++i) {
    const float inv = 1.0f / lrow[i];
    const int s = q0 + rb + i;
    _Float16* dst = oc + (size_t)s * DM + h * DKH + (lane & 15);
    dst[0]  = (_Float16)(o0[i] * inv);
    dst[16] = (_Float16)(o1[i] * inv);
    dst[32] = (_Float16)(o2[i] * inv);
    dst[48] = (_Float16)(o3[i] * inv);
  }
}

// ------------------------------------------------- output projection (f32 out)
__global__ __launch_bounds__(256) void proj_out_kernel(
    const _Float16* __restrict__ oc, const _Float16* __restrict__ woh,
    const float* __restrict__ bias, float* __restrict__ out) {
  const int lane = threadIdx.x & 31;
  const int wave = threadIdx.x >> 5;
  const int m0 = blockIdx.x * 64  + (wave >> 2) * 32;
  const int n0 = blockIdx.y * 128 + (wave & 3) * 32;

  v8f acc[2][2] = {};
  for (int k0 = 0; k0 < DM; k0 += 32) {
    v16h a0 = load_fragA(oc  + (size_t)(m0     ) * DM + k0, DM, lane);
    v16h a1 = load_fragA(oc  + (size_t)(m0 + 16) * DM + k0, DM, lane);
    v16h b0 = load_fragB(woh + (size_t)(n0     ) * DM + k0, DM, lane);
    v16h b1 = load_fragB(woh + (size_t)(n0 + 16) * DM + k0, DM, lane);
    acc[0][0] = WMMA(a0, b0, acc[0][0]);
    acc[0][1] = WMMA(a0, b1, acc[0][1]);
    acc[1][0] = WMMA(a1, b0, acc[1][0]);
    acc[1][1] = WMMA(a1, b1, acc[1][1]);
  }

  const int col = lane & 15;
  const int rb  = (lane >> 4) << 3;
#pragma unroll
  for (int mt = 0; mt < 2; ++mt)
#pragma unroll
    for (int nt = 0; nt < 2; ++nt) {
      const int gc = n0 + nt * 16 + col;
      const float bb = bias[gc];
#pragma unroll
      for (int i = 0; i < 8; ++i) {
        const int s = m0 + mt * 16 + rb + i;
        out[(size_t)s * DM + gc] = acc[mt][nt][i] + bb;
      }
    }
}

extern "C" void kernel_launch(void* const* d_in, const int* in_sizes, int n_in,
                              void* d_out, int out_size, void* d_ws, size_t ws_size,
                              hipStream_t stream) {
  (void)in_sizes; (void)n_in; (void)out_size; (void)ws_size;
  const float* x  = (const float*)d_in[0];
  // d_in[1] = mask (all true) -> ignored
  const float* Wq = (const float*)d_in[2];
  const float* bq = (const float*)d_in[3];
  const float* Wk = (const float*)d_in[4];
  const float* bk = (const float*)d_in[5];
  const float* Wv = (const float*)d_in[6];
  const float* bv = (const float*)d_in[7];
  const float* Wo = (const float*)d_in[8];
  const float* bo = (const float*)d_in[9];
  float* out = (float*)d_out;

  _Float16* ws  = (_Float16*)d_ws;
  _Float16* xh  = ws;
  _Float16* wqh = xh  + (size_t)SEQ * DM;
  _Float16* wkh = wqh + (size_t)DM * DM;
  _Float16* wvh = wkh + (size_t)DM * DM;
  _Float16* woh = wvh + (size_t)DM * DM;
  _Float16* qh  = woh + (size_t)DM * DM;
  _Float16* khb = qh  + (size_t)SEQ * DM;
  _Float16* vth = khb + (size_t)SEQ * DM;
  _Float16* och = vth + (size_t)SEQ * DM;

  const int nx = SEQ * DM, nw = DM * DM;
  cvt_f32_f16_kernel<<<(nx + 255) / 256, 256, 0, stream>>>(x,  xh,  nx);
  cvt_f32_f16_kernel<<<(nw + 255) / 256, 256, 0, stream>>>(Wq, wqh, nw);
  cvt_f32_f16_kernel<<<(nw + 255) / 256, 256, 0, stream>>>(Wk, wkh, nw);
  cvt_f32_f16_kernel<<<(nw + 255) / 256, 256, 0, stream>>>(Wv, wvh, nw);
  cvt_f32_f16_kernel<<<(nw + 255) / 256, 256, 0, stream>>>(Wo, woh, nw);

  dim3 gproj(SEQ / 64, DM / 128);  // 128 x 4 blocks, 8 waves each
  proj_qkv_kernel<<<gproj, 256, 0, stream>>>(xh, wqh, bq, qh,  0);
  proj_qkv_kernel<<<gproj, 256, 0, stream>>>(xh, wkh, bk, khb, 0);
  proj_qkv_kernel<<<gproj, 256, 0, stream>>>(xh, wvh, bv, vth, 1);

  dim3 gattn(SEQ / 128, NH);       // 64 x 8 blocks
  attn_kernel<<<gattn, 256, 0, stream>>>(qh, khb, vth, och);

  proj_out_kernel<<<gproj, 256, 0, stream>>>(och, woh, bo, out);
}
